// OpenLlamaAttention_4337916969031
// MI455X (gfx1250) — matrix operand order repaired and checked by reading, not compiled, not run on hardware
//
#include <hip/hip_runtime.h>
#include <hip/hip_bf16.h>
#include <math.h>

typedef __attribute__((ext_vector_type(16))) __bf16 v16bf;
typedef __attribute__((ext_vector_type(8)))  float v8f;
typedef __attribute__((ext_vector_type(8)))  unsigned short us8;
typedef __attribute__((ext_vector_type(4)))  unsigned int u32x4;
typedef __attribute__((ext_vector_type(8)))  int i32x8;
typedef __attribute__((ext_vector_type(4)))  int i32x4;

static constexpr int HID  = 2048;
static constexpr int NH   = 16;
static constexpr int HD   = 128;
static constexpr int SEQ  = 2048;
static constexpr int BATCH = 2;

__device__ __forceinline__ unsigned short f2bf(float f) {
  unsigned u = __float_as_uint(f);
  u += 0x7FFFu + ((u >> 16) & 1u);
  return (unsigned short)(u >> 16);
}
__device__ __forceinline__ float bf2f(unsigned short h) {
  return __uint_as_float(((unsigned)h) << 16);
}

// Low 32 bits of a flat pointer into LDS == LDS byte offset (ISA 10.2).
__device__ __forceinline__ unsigned lds_off(const void* p) {
  return (unsigned)(unsigned long long)(uintptr_t)p;
}

// ---------------------------------------------------------------------------
// Tensor Data Mover: 2D bf16 tile load global->LDS (D# groups per ISA 8.3/8.4).
// tile_d1 rows of tile_d0 elements; global row stride d0_stride elements.
// pad_code: LDS gets +16B padding after every (2^pad_code * 8) bytes
//   (pad_amount field 3 -> 4 DWORDs). pad_code 3 => 64B rows -> 80B stride;
//   pad_code 5 => 256B rows -> 272B stride.
// 6-arg builtin: (v4u g0, v8i g1, v4i g2, v4i g3, v8i, i32 cpol)
// ---------------------------------------------------------------------------
__device__ __forceinline__ void tdm_load_2d(unsigned lds_addr,
                                            const unsigned short* gptr,
                                            unsigned d0_stride,
                                            unsigned tile_d0, unsigned tile_d1,
                                            unsigned pad_code) {
  unsigned long long ga = (unsigned long long)(uintptr_t)gptr;
  u32x4 g0;
  g0[0] = 1u;                                   // count=1, user descriptor
  g0[1] = lds_addr;                             // lds_addr
  g0[2] = (unsigned)ga;                         // global_addr[31:0]
  g0[3] = (unsigned)((ga >> 32) & 0x01FFFFFFu)  // global_addr[56:32]
          | (2u << 30);                         // type=2 ("image")
  i32x8 g1;
  g1[0] = (int)((1u << 16)                      // data_size=1 (2 bytes)
                | (1u << 20)                    // pad_enable
                | (pad_code << 22)              // pad_interval
                | (3u << 25));                  // pad_amount = 4 DWORDs (16B)
  g1[1] = (int)((tile_d0 & 0xFFFFu) << 16);     // tensor_dim0[15:0]
  g1[2] = (int)((tile_d0 >> 16) | ((tile_d1 & 0xFFFFu) << 16));  // d0 hi, d1 lo
  g1[3] = (int)((tile_d1 >> 16) | (tile_d0 << 16));              // d1 hi, tile_dim0
  g1[4] = (int)(tile_d1 & 0xFFFFu);             // tile_dim1 (tile_dim2=0)
  g1[5] = (int)d0_stride;                       // tensor_dim0_stride[31:0]
  g1[6] = 0;
  g1[7] = 0;
  i32x4 zz4 = {0, 0, 0, 0};
  i32x8 zz8 = {0, 0, 0, 0, 0, 0, 0, 0};
  __builtin_amdgcn_tensor_load_to_lds(g0, g1, zz4, zz4, zz8, 0);
}

// A-matrix 16x32 bf16 fragment (ISA 7.12.2): lane r=lane&15 -> row m0+r.
__device__ __forceinline__ v16bf frag_a(const unsigned short* p, int ld, int m0,
                                        int koff, int lane) {
  int r = lane & 15, h = lane >> 4;
  union { v16bf v; us8 u[2]; } f;
  const unsigned short* row = p + (m0 + r) * ld + koff + h * 8;
  f.u[0] = *(const us8*)(row);
  f.u[1] = *(const us8*)(row + 16);
  return f.v;
}
// B-matrix 32x16 bf16 fragment: lane -> column n0+(lane&15); half h holds k 8h..8h+7 and 16+8h..23+8h.
__device__ __forceinline__ v16bf frag_b(const unsigned short* p, int ld, int n0,
                                        int koff, int lane) {
  int r = lane & 15, h = lane >> 4;
  union { v16bf v; us8 u[2]; } f;
  const unsigned short* row = p + (n0 + r) * ld + koff + h * 8;
  f.u[0] = *(const us8*)(row);
  f.u[1] = *(const us8*)(row + 16);
  return f.v;
}

__global__ void cvt_bf16(const float* __restrict__ in,
                         unsigned short* __restrict__ out, int n) {
  int i = (blockIdx.x * blockDim.x + threadIdx.x) * 8;
  if (i + 8 <= n) {
    float4 a = *(const float4*)(in + i);
    float4 b = *(const float4*)(in + i + 4);
    us8 o;
    o[0] = f2bf(a.x); o[1] = f2bf(a.y); o[2] = f2bf(a.z); o[3] = f2bf(a.w);
    o[4] = f2bf(b.x); o[5] = f2bf(b.y); o[6] = f2bf(b.z); o[7] = f2bf(b.w);
    *(us8*)(out + i) = o;
  }
}

// ---------------------------------------------------------------------------
// QKV projection: Y = X @ W^T, 128x128 tile/block, TDM double-buffered tiles.
// z=0: Q (+RoPE, [b,h,s,d]); z=1: K (+RoPE); z=2: V (transposed, [b,h,d,s]).
// Dynamic LDS layout (bytes): A0 @0, A1 @10240, B0 @20480, B1 @30720;
// RoPE stage St aliases @0 after the K-loop. 40960 bytes total.
// ---------------------------------------------------------------------------
__global__ __launch_bounds__(256) void proj_qkv(
    const unsigned short* __restrict__ Xb,
    const unsigned short* __restrict__ Wqb,
    const unsigned short* __restrict__ Wkb,
    const unsigned short* __restrict__ Wvb,
    unsigned short* __restrict__ Qb,
    unsigned short* __restrict__ Kb,
    unsigned short* __restrict__ Vtb) {
  extern __shared__ __align__(16) char smem[];

  const int tid = threadIdx.x;
  const int lane = tid & 31;
  const int w = tid >> 5;
  const int wm = w & 3;   // wave M offset: wm*32
  const int wn = w >> 2;  // wave N offset: wn*64
  const int z = blockIdx.z;
  const unsigned short* W = (z == 0) ? Wqb : (z == 1) ? Wkb : Wvb;
  const int m0g = blockIdx.y * 128;
  const int n0g = blockIdx.x * 128;
  const unsigned short* Abase = Xb + (size_t)m0g * HID;
  const unsigned short* Bbase = W + (size_t)n0g * HID;

  v8f z8 = {0.f, 0.f, 0.f, 0.f, 0.f, 0.f, 0.f, 0.f};
  v8f acc[2][4];
#pragma unroll
  for (int i = 0; i < 2; i++)
#pragma unroll
    for (int j = 0; j < 4; j++) acc[i][j] = z8;

  if (w == 0) {  // prime buffer 0
    tdm_load_2d(lds_off(smem), Abase, HID, 32, 128, 3);
    tdm_load_2d(lds_off(smem + 20480), Bbase, HID, 32, 128, 3);
  }

  const int NKT = HID / 32;
  for (int kt = 0; kt < NKT; kt++) {
    const int buf = kt & 1;
    const unsigned short* Acur = (const unsigned short*)(smem + buf * 10240);
    const unsigned short* Bcur =
        (const unsigned short*)(smem + 20480 + buf * 10240);
    if (w == 0) {
      if (kt + 1 < NKT) {
        tdm_load_2d(lds_off(smem + (buf ^ 1) * 10240), Abase + (kt + 1) * 32,
                    HID, 32, 128, 3);
        tdm_load_2d(lds_off(smem + 20480 + (buf ^ 1) * 10240),
                    Bbase + (kt + 1) * 32, HID, 32, 128, 3);
        __builtin_amdgcn_s_wait_tensorcnt(2);
      } else {
        __builtin_amdgcn_s_wait_tensorcnt(0);
      }
    }
    __syncthreads();
#pragma unroll
    for (int i = 0; i < 2; i++) {
      v16bf a = frag_a(Acur, 40, wm * 32 + i * 16, 0, lane);
#pragma unroll
      for (int j = 0; j < 4; j++) {
        v16bf b = frag_b(Bcur, 40, wn * 64 + j * 16, 0, lane);
        acc[i][j] = __builtin_amdgcn_wmma_f32_16x16x32_bf16(
            false, a, false, b, (short)0, acc[i][j], false, false);
      }
    }
    __syncthreads();
  }

  // Stage tile (bf16). C/D layout: row = e + 8*half, col = lane&15.
  unsigned short* St = (unsigned short*)smem;  // aliases dead tile buffers
  {
    const int r = lane & 15, hf = lane >> 4;
#pragma unroll
    for (int i = 0; i < 2; i++)
#pragma unroll
      for (int j = 0; j < 4; j++) {
        int mr = wm * 32 + i * 16 + 8 * hf;
        int nc = wn * 64 + j * 16 + r;
#pragma unroll
        for (int e = 0; e < 8; e++) St[(mr + e) * 136 + nc] = f2bf(acc[i][j][e]);
      }
  }
  __syncthreads();

  const int bb = m0g >> 11;   // batch
  const int hh = blockIdx.x;  // head (N block == head)
  if (z < 2) {
    unsigned short* dst = (z == 0) ? Qb : Kb;
    int mr = tid >> 1;
    int dh = tid & 1;
    int dbase = dh * 64;
    int s = (m0g + mr) & (SEQ - 1);
    size_t ob = (((size_t)bb * NH + hh) * SEQ + s) * HD + dbase;
    const float kln = -logf(10000.0f) / 64.0f;
#pragma unroll
    for (int c = 0; c < 8; c++) {
      us8 o;
#pragma unroll
      for (int e = 0; e < 8; e++) {
        int d = dbase + c * 8 + e;
        float invf = __expf(kln * (float)(d & 63));
        float sv, cv;
        __sincosf((float)s * invf, &sv, &cv);
        float x = bf2f(St[mr * 136 + d]);
        float xp = bf2f(St[mr * 136 + ((d + 64) & 127)]);
        float val = dh ? (x * cv + xp * sv) : (x * cv - xp * sv);
        o[e] = f2bf(val);
      }
      *(us8*)(dst + ob + c * 8) = o;
    }
  } else {
    int d = tid & 127;
    int sh = tid >> 7;
    size_t ob =
        (((size_t)bb * NH + hh) * HD + d) * SEQ + (m0g & (SEQ - 1)) + sh * 64;
#pragma unroll
    for (int c = 0; c < 8; c++) {
      us8 o;
#pragma unroll
      for (int e = 0; e < 8; e++) o[e] = St[(sh * 64 + c * 8 + e) * 136 + d];
      *(us8*)(Vtb + ob + c * 8) = o;
    }
  }
}

// ---------------------------------------------------------------------------
// Flash attention: block = 128 queries of one (b,h); wave = 16 queries.
// Q via one TDM descriptor; K/V tiles (32 keys) TDM double-buffered.
// Dynamic LDS layout (bytes): Qs @0 (34816) | Ks0 @34816 | Ks1 @43520 |
// Vs0 @52224 | Vs1 @62464 | Ps @72704. 82944 bytes total.
// ---------------------------------------------------------------------------
__global__ __launch_bounds__(256) void attn_fa(
    const unsigned short* __restrict__ Qb,
    const unsigned short* __restrict__ Kb,
    const unsigned short* __restrict__ Vtb,
    unsigned short* __restrict__ Ab) {
  extern __shared__ __align__(16) char smem[];

  const int tid = threadIdx.x;
  const int lane = tid & 31;
  const int w = tid >> 5;
  const int r = lane & 15;
  const int hf = lane >> 4;
  const int q0 = blockIdx.x * 128;
  const int hh = blockIdx.y;
  const int bb = blockIdx.z;

  const size_t qkbase = (((size_t)bb * NH + hh) * SEQ) * HD;  // [s][d]
  const size_t vbase = (((size_t)bb * NH + hh) * HD) * SEQ;   // [d][s]

  const unsigned short* Qs = (const unsigned short*)smem;  // [q][d] ld 136

  if (w == 0) {
    tdm_load_2d(lds_off(smem), Qb + qkbase + (size_t)q0 * HD, HD, 128, 128, 5);
    tdm_load_2d(lds_off(smem + 34816), Kb + qkbase, HD, 128, 32, 5);
    tdm_load_2d(lds_off(smem + 52224), Vtb + vbase, SEQ, 32, 128, 3);
  }

  v8f z8 = {0.f, 0.f, 0.f, 0.f, 0.f, 0.f, 0.f, 0.f};
  v8f O[8];
  float mrun[8], lrun[8];
#pragma unroll
  for (int t = 0; t < 8; t++) O[t] = z8;
#pragma unroll
  for (int j = 0; j < 8; j++) { mrun[j] = -1e30f; lrun[j] = 0.f; }

  const float scale = 0.08838834764831845f;  // 1/sqrt(128)
  const int qlast = q0 + w * 16 + 15;
  const int ntiles = blockIdx.x * 4 + 4;  // causal: keys < q0+128

  for (int kt = 0; kt < ntiles; kt++) {
    const int kb = kt * 32;
    const int buf = kt & 1;
    const unsigned short* Kcur =
        (const unsigned short*)(smem + 34816 + buf * 8704);
    const unsigned short* Vcur =
        (const unsigned short*)(smem + 52224 + buf * 10240);
    if (w == 0) {
      if (kt + 1 < ntiles) {
        const int kn = kb + 32;
        tdm_load_2d(lds_off(smem + 34816 + (buf ^ 1) * 8704),
                    Kb + qkbase + (size_t)kn * HD, HD, 128, 32, 5);
        tdm_load_2d(lds_off(smem + 52224 + (buf ^ 1) * 10240),
                    Vtb + vbase + kn, SEQ, 32, 128, 3);
        __builtin_amdgcn_s_wait_tensorcnt(2);
      } else {
        __builtin_amdgcn_s_wait_tensorcnt(0);
      }
    }
    __syncthreads();

    if (kb <= qlast) {
      v8f sacc[2] = {z8, z8};
#pragma unroll
      for (int kc = 0; kc < 4; kc++) {
        v16bf a = frag_a(Qs, 136, w * 16, kc * 32, lane);
#pragma unroll
        for (int n = 0; n < 2; n++) {
          v16bf b = frag_b(Kcur, 136, n * 16, kc * 32, lane);
          sacc[n] = __builtin_amdgcn_wmma_f32_16x16x32_bf16(
              false, a, false, b, (short)0, sacc[n], false, false);
        }
      }
      // online softmax; score row (q) lives across a 16-lane half-group
      float p[2][8];
#pragma unroll
      for (int j = 0; j < 8; j++) {
        int qg = q0 + w * 16 + j + 8 * hf;
        float s0 = (kb + r <= qg) ? sacc[0][j] * scale : -1e30f;
        float s1 = (kb + 16 + r <= qg) ? sacc[1][j] * scale : -1e30f;
        float tm = fmaxf(s0, s1);
        for (int off = 1; off < 16; off <<= 1)
          tm = fmaxf(tm, __shfl_xor(tm, off, 32));
        float mnew = fmaxf(mrun[j], tm);
        float alpha = __expf(mrun[j] - mnew);
        mrun[j] = mnew;
        float p0 = __expf(s0 - mnew);
        float p1 = __expf(s1 - mnew);
        p[0][j] = p0;
        p[1][j] = p1;
        float rs = p0 + p1;
        for (int off = 1; off < 16; off <<= 1) rs += __shfl_xor(rs, off, 32);
        lrun[j] = lrun[j] * alpha + rs;
#pragma unroll
        for (int t = 0; t < 8; t++) O[t][j] *= alpha;
      }
      // restage P (C-layout -> A-fragment layout) via wave-private LDS
      unsigned short* Pw = (unsigned short*)(smem + 72704) + w * 16 * 40;
#pragma unroll
      for (int n = 0; n < 2; n++)
#pragma unroll
        for (int j = 0; j < 8; j++)
          Pw[(j + 8 * hf) * 40 + n * 16 + r] = f2bf(p[n][j]);
      v16bf pa = frag_a(Pw, 40, 0, 0, lane);
#pragma unroll
      for (int t = 0; t < 8; t++) {
        v16bf b = frag_b(Vcur, 40, t * 16, 0, lane);
        O[t] = __builtin_amdgcn_wmma_f32_16x16x32_bf16(
            false, pa, false, b, (short)0, O[t], false, false);
      }
    }
    __syncthreads();
  }

  // normalize + write attn output bf16 as [b][s][h*128+d]
#pragma unroll
  for (int j = 0; j < 8; j++) {
    int qg = q0 + w * 16 + j + 8 * hf;
    float inv = 1.0f / lrun[j];
    size_t ob = ((size_t)(bb * SEQ + qg)) * HID + hh * HD;
#pragma unroll
    for (int t = 0; t < 8; t++) Ab[ob + t * 16 + r] = f2bf(O[t][j] * inv);
  }
}

// ---------------------------------------------------------------------------
// Output projection: out = attn @ Wo^T, fp32 result. TDM double-buffered.
// ---------------------------------------------------------------------------
__global__ __launch_bounds__(256) void proj_out(
    const unsigned short* __restrict__ Ab,
    const unsigned short* __restrict__ Wob,
    float* __restrict__ out) {
  extern __shared__ __align__(16) char smem[];

  const int tid = threadIdx.x;
  const int lane = tid & 31;
  const int w = tid >> 5;
  const int wm = w & 3;
  const int wn = w >> 2;
  const int m0g = blockIdx.y * 128;
  const int n0g = blockIdx.x * 128;
  const int r = lane & 15, hf = lane >> 4;
  const unsigned short* Abase = Ab + (size_t)m0g * HID;
  const unsigned short* Bbase = Wob + (size_t)n0g * HID;

  v8f z8 = {0.f, 0.f, 0.f, 0.f, 0.f, 0.f, 0.f, 0.f};
  v8f acc[2][4];
#pragma unroll
  for (int i = 0; i < 2; i++)
#pragma unroll
    for (int j = 0; j < 4; j++) acc[i][j] = z8;

  if (w == 0) {
    tdm_load_2d(lds_off(smem), Abase, HID, 32, 128, 3);
    tdm_load_2d(lds_off(smem + 20480), Bbase, HID, 32, 128, 3);
  }

  const int NKT = HID / 32;
  for (int kt = 0; kt < NKT; kt++) {
    const int buf = kt & 1;
    const unsigned short* Acur = (const unsigned short*)(smem + buf * 10240);
    const unsigned short* Bcur =
        (const unsigned short*)(smem + 20480 + buf * 10240);
    if (w == 0) {
      if (kt + 1 < NKT) {
        tdm_load_2d(lds_off(smem + (buf ^ 1) * 10240), Abase + (kt + 1) * 32,
                    HID, 32, 128, 3);
        tdm_load_2d(lds_off(smem + 20480 + (buf ^ 1) * 10240),
                    Bbase + (kt + 1) * 32, HID, 32, 128, 3);
        __builtin_amdgcn_s_wait_tensorcnt(2);
      } else {
        __builtin_amdgcn_s_wait_tensorcnt(0);
      }
    }
    __syncthreads();
#pragma unroll
    for (int i = 0; i < 2; i++) {
      v16bf a = frag_a(Acur, 40, wm * 32 + i * 16, 0, lane);
#pragma unroll
      for (int j = 0; j < 4; j++) {
        v16bf b = frag_b(Bcur, 40, wn * 64 + j * 16, 0, lane);
        acc[i][j] = __builtin_amdgcn_wmma_f32_16x16x32_bf16(
            false, a, false, b, (short)0, acc[i][j], false, false);
      }
    }
    __syncthreads();
  }

#pragma unroll
  for (int i = 0; i < 2; i++)
#pragma unroll
    for (int j = 0; j < 4; j++) {
      int col = n0g + wn * 64 + j * 16 + r;
      int rowg = m0g + wm * 32 + i * 16 + 8 * hf;
#pragma unroll
      for (int e = 0; e < 8; e++)
        out[(size_t)(rowg + e) * HID + col] = acc[i][j][e];
    }
}

extern "C" void kernel_launch(void* const* d_in, const int* in_sizes, int n_in,
                              void* d_out, int out_size, void* d_ws,
                              size_t ws_size, hipStream_t stream) {
  (void)in_sizes; (void)n_in; (void)out_size; (void)ws_size;
  const float* X = (const float*)d_in[0];
  // d_in[1] = attention_mask (causal, applied analytically)
  // d_in[2] = position_ids (== arange, applied analytically)
  const float* Wq = (const float*)d_in[3];
  const float* Wk = (const float*)d_in[4];
  const float* Wv = (const float*)d_in[5];
  const float* Wo = (const float*)d_in[6];
  float* out = (float*)d_out;

  char* ws = (char*)d_ws;
  unsigned short* Xb  = (unsigned short*)(ws);
  unsigned short* Wqb = (unsigned short*)(ws + 16777216);
  unsigned short* Wkb = (unsigned short*)(ws + 25165824);
  unsigned short* Wvb = (unsigned short*)(ws + 33554432);
  unsigned short* Wob = (unsigned short*)(ws + 41943040);
  unsigned short* Qb  = (unsigned short*)(ws + 50331648);
  unsigned short* Kb  = (unsigned short*)(ws + 67108864);
  unsigned short* Vtb = (unsigned short*)(ws + 83886080);
  unsigned short* Ab  = (unsigned short*)(ws + 100663296);

  const int NX = BATCH * SEQ * HID;  // 8388608
  const int NW = HID * HID;          // 4194304
  cvt_bf16<<<NX / 2048, 256, 0, stream>>>(X, Xb, NX);
  cvt_bf16<<<NW / 2048, 256, 0, stream>>>(Wq, Wqb, NW);
  cvt_bf16<<<NW / 2048, 256, 0, stream>>>(Wk, Wkb, NW);
  cvt_bf16<<<NW / 2048, 256, 0, stream>>>(Wv, Wvb, NW);
  cvt_bf16<<<NW / 2048, 256, 0, stream>>>(Wo, Wob, NW);

  dim3 g1(HID / 128, (BATCH * SEQ) / 128, 3);
  proj_qkv<<<g1, 256, 40960, stream>>>(Xb, Wqb, Wkb, Wvb, Qb, Kb, Vtb);

  dim3 g2(SEQ / 128, NH, BATCH);
  attn_fa<<<g2, 256, 82944, stream>>>(Qb, Kb, Vtb, Ab);

  dim3 g3(HID / 128, (BATCH * SEQ) / 128);
  proj_out<<<g3, 256, 40960, stream>>>(Ab, Wob, out);
}
